// TSSGCF_5153960755405
// MI455X (gfx1250) — compile-verified, matrix operand order checked by reading, not run on hardware
//
#include <hip/hip_runtime.h>
#include <stdint.h>

typedef __attribute__((ext_vector_type(16))) __bf16        v16bf;
typedef __attribute__((ext_vector_type(8)))  float         v8f;
typedef __attribute__((ext_vector_type(8)))  unsigned int  v8u;

#define DK 384          // text feature dim
#define DN 64           // embedding dim
#define LDS_STRIDE 392  // 384 + 8 pad (ushorts) -> 784B row stride, bank-conflict free

// ---- fp32 -> bf16 (round to nearest even) ----
__device__ __forceinline__ unsigned pk_bf16(float a, float b) {
  unsigned ua = __float_as_uint(a);
  unsigned ub = __float_as_uint(b);
  ua += 0x7FFFu + ((ua >> 16) & 1u);
  ub += 0x7FFFu + ((ub >> 16) & 1u);
  return (ua >> 16) | (ub & 0xFFFF0000u);
}
__device__ __forceinline__ unsigned short cv_bf16(float a) {
  unsigned ua = __float_as_uint(a);
  ua += 0x7FFFu + ((ua >> 16) & 1u);
  return (unsigned short)(ua >> 16);
}

// ---------------- elementwise helpers ----------------
__global__ void k_zero(float4* __restrict__ p, long n4) {
  long i = (long)blockIdx.x * blockDim.x + threadIdx.x;
  if (i < n4) p[i] = make_float4(0.f, 0.f, 0.f, 0.f);
}
__global__ void k_copy(float4* __restrict__ dst, const float4* __restrict__ src, long n4) {
  long i = (long)blockIdx.x * blockDim.x + threadIdx.x;
  if (i < n4) dst[i] = src[i];
}
__global__ void k_add(float4* __restrict__ dst, const float4* __restrict__ src, long n4) {
  long i = (long)blockIdx.x * blockDim.x + threadIdx.x;
  if (i < n4) {
    float4 d = dst[i], s = src[i];
    d.x += s.x; d.y += s.y; d.z += s.z; d.w += s.w;
    dst[i] = d;
  }
}

// ---------------- SpMM scatter: y[row[e],:] += val[e] * x[col[e],:] ----------------
// 16 lanes per edge, float4 (b128) gathers, f32 atomics (L2-side).
__global__ void k_spmm(const int* __restrict__ row, const int* __restrict__ col,
                       const float* __restrict__ val, const float* __restrict__ x,
                       float* __restrict__ y, int E) {
  long tid = (long)blockIdx.x * blockDim.x + threadIdx.x;
  int e = (int)(tid >> 4);
  if (e >= E) return;
  int c = ((int)tid & 15) << 2;
  int r = row[e], s = col[e];
  float v = val[e];
  const float4 xv = *reinterpret_cast<const float4*>(x + (size_t)s * DN + c);
  float* yp = y + (size_t)r * DN + c;
  atomicAdd(yp + 0, v * xv.x);
  atomicAdd(yp + 1, v * xv.y);
  atomicAdd(yp + 2, v * xv.z);
  atomicAdd(yp + 3, v * xv.w);
}

// ---------------- in-place: g = 0.5 * l2norm(acc/4) = acc * 0.5/max(||acc||, 4eps) ----------------
__global__ void k_graph_norm(float* __restrict__ g, int N) {
  int r    = blockIdx.x * (blockDim.x >> 5) + (threadIdx.x >> 5);
  int lane = threadIdx.x & 31;
  if (r >= N) return;
  float2 x = *reinterpret_cast<float2*>(g + (size_t)r * DN + lane * 2);
  float s = x.x * x.x + x.y * x.y;
  #pragma unroll
  for (int off = 1; off < 32; off <<= 1) s += __shfl_xor(s, off, 32);
  float rs = 0.5f / fmaxf(sqrtf(s), 4e-12f);
  x.x *= rs; x.y *= rs;
  *reinterpret_cast<float2*>(g + (size_t)r * DN + lane * 2) = x;
}

// ---------------- text projection: out += 0.5 * l2norm(A @ W + b) ----------------
// 4 waves/block; each wave owns a 16-row block, all 64 output cols (4 WMMA N-tiles).
__global__ void __launch_bounds__(128) k_text_wmma(
    const float* __restrict__ A,     // N x 384 fp32
    const float* __restrict__ W,     // 384 x 64 fp32
    const float* __restrict__ bias,  // 64
    float* __restrict__ out,         // N x 64 (holds 0.5*g_norm already)
    int N) {
  __shared__ unsigned short lds[DN * LDS_STRIDE];  // W^T in bf16, padded rows

  const int tid = threadIdx.x;
  // Cooperative W -> LDS (transposed, bf16). Contiguous 16-bit LDS stores.
  for (int i = tid; i < DN * DK; i += 128) {
    int n = i / DK;
    int k = i - n * DK;
    lds[n * LDS_STRIDE + k] = cv_bf16(W[(size_t)k * DN + n]);
  }
  __syncthreads();

  const int wv   = tid >> 5;
  const int lane = tid & 31;
  const int m0   = (blockIdx.x * 4 + wv) * 16;
  if (m0 >= N) return;
  const int ml   = lane & 15;
  const int hi   = lane >> 4;
  const int koff = hi * 8;          // ISA 16-bit A/B layout: half-wave K offset
  int m = m0 + ml;
  if (m >= N) m = N - 1;            // clamp (N is a multiple of 16 in practice)
  const float* arow = A + (size_t)m * DK;

  v8f c[4];
  #pragma unroll
  for (int t = 0; t < 4; t++) {
    float bv = bias[t * 16 + ml];
    #pragma unroll
    for (int i = 0; i < 8; i++) c[t][i] = bv;
  }

  for (int k = 0; k < DK; k += 32) {
    // A fragment: K = [k+koff, k+koff+8) U [k+16+koff, k+16+koff+8)
    const float4* pa0 = reinterpret_cast<const float4*>(arow + k + koff);
    const float4* pa1 = reinterpret_cast<const float4*>(arow + k + 16 + koff);
    float4 a0 = pa0[0], a1 = pa0[1];
    float4 a2 = pa1[0], a3 = pa1[1];
    v8u au;
    au[0] = pk_bf16(a0.x, a0.y); au[1] = pk_bf16(a0.z, a0.w);
    au[2] = pk_bf16(a1.x, a1.y); au[3] = pk_bf16(a1.z, a1.w);
    au[4] = pk_bf16(a2.x, a2.y); au[5] = pk_bf16(a2.z, a2.w);
    au[6] = pk_bf16(a3.x, a3.y); au[7] = pk_bf16(a3.z, a3.w);
    v16bf a = __builtin_bit_cast(v16bf, au);

    #pragma unroll
    for (int t = 0; t < 4; t++) {
      int n = t * 16 + ml;
      const uint4 q0 = *reinterpret_cast<const uint4*>(&lds[n * LDS_STRIDE + k + koff]);
      const uint4 q1 = *reinterpret_cast<const uint4*>(&lds[n * LDS_STRIDE + k + 16 + koff]);
      v8u bu;
      bu[0] = q0.x; bu[1] = q0.y; bu[2] = q0.z; bu[3] = q0.w;
      bu[4] = q1.x; bu[5] = q1.y; bu[6] = q1.z; bu[7] = q1.w;
      v16bf b = __builtin_bit_cast(v16bf, bu);
      c[t] = __builtin_amdgcn_wmma_f32_16x16x32_bf16(
          false, a, false, b, (short)0, c[t], false, false);
    }
  }

  // Row L2-norm: row (m0 + v + 8*hi) lives in c[*][v] across the 16 lanes of this half.
  float ssq[8];
  #pragma unroll
  for (int v = 0; v < 8; v++) {
    float s = 0.f;
    #pragma unroll
    for (int t = 0; t < 4; t++) { float e = c[t][v]; s += e * e; }
    #pragma unroll
    for (int off = 1; off < 16; off <<= 1) s += __shfl_xor(s, off, 32);
    ssq[v] = s;
  }

  #pragma unroll
  for (int v = 0; v < 8; v++) {
    int r = m0 + v + 8 * hi;
    if (r >= N) continue;
    float rs = 0.5f / fmaxf(sqrtf(ssq[v]), 1e-12f);
    #pragma unroll
    for (int t = 0; t < 4; t++) {
      size_t idx = (size_t)r * DN + t * 16 + ml;
      out[idx] = out[idx] + c[t][v] * rs;
    }
  }
}

// ---------------- host launcher ----------------
extern "C" void kernel_launch(void* const* d_in, const int* in_sizes, int n_in,
                              void* d_out, int out_size, void* d_ws, size_t ws_size,
                              hipStream_t stream) {
  (void)n_in; (void)out_size; (void)ws_size;

  const int*   m_row = (const int*)d_in[0];
  const int*   m_col = (const int*)d_in[1];
  const float* m_val = (const float*)d_in[2];
  const int*   a_row = (const int*)d_in[3];
  const int*   a_col = (const int*)d_in[4];
  const float* a_val = (const float*)d_in[5];
  const float* m_txt = (const float*)d_in[6];
  const float* a_txt = (const float*)d_in[7];
  const float* m_emb = (const float*)d_in[8];
  const float* a_emb = (const float*)d_in[9];
  const float* W     = (const float*)d_in[10];
  const float* bias  = (const float*)d_in[11];

  const int E_m = in_sizes[0];
  const int E_a = in_sizes[3];
  const int N_m = in_sizes[8] / DN;
  const int N_a = in_sizes[9] / DN;
  const int N_mx = (N_m > N_a) ? N_m : N_a;

  float* out_m = (float*)d_out;
  float* out_a = out_m + (size_t)N_m * DN;
  float* ws0 = (float*)d_ws;
  float* ws1 = ws0 + (size_t)N_mx * DN;
  float* buf[2] = {ws0, ws1};

  struct G { const int* row; const int* col; const float* val;
             const float* emb; int N; int E; float* out; };
  G gs[2] = {{m_row, m_col, m_val, m_emb, N_m, E_m, out_m},
             {a_row, a_col, a_val, a_emb, N_a, E_a, out_a}};

  const int blk = 256;
  for (int gi = 0; gi < 2; gi++) {
    G& g = gs[gi];
    long n4 = ((long)g.N * DN) >> 2;
    unsigned gew = (unsigned)((n4 + blk - 1) / blk);

    // acc (in d_out) = emb
    k_copy<<<dim3(gew), blk, 0, stream>>>((float4*)g.out, (const float4*)g.emb, n4);

    const float* cur = g.emb;
    for (int l = 0; l < 3; l++) {
      float* nxt = buf[l & 1];
      k_zero<<<dim3(gew), blk, 0, stream>>>((float4*)nxt, n4);
      long nthreads = (long)g.E * 16;
      unsigned gsp = (unsigned)((nthreads + blk - 1) / blk);
      k_spmm<<<dim3(gsp), blk, 0, stream>>>(g.row, g.col, g.val, cur, nxt, g.E);
      k_add<<<dim3(gew), blk, 0, stream>>>((float4*)g.out, (const float4*)nxt, n4);
      cur = nxt;
    }
    unsigned gnm = (unsigned)((g.N + (blk / 32) - 1) / (blk / 32));
    k_graph_norm<<<dim3(gnm), blk, 0, stream>>>(g.out, g.N);
  }

  // text projection + combine (WMMA)
  unsigned gtm = (unsigned)((N_m + 63) / 64);
  unsigned gta = (unsigned)((N_a + 63) / 64);
  k_text_wmma<<<dim3(gtm), 128, 0, stream>>>(m_txt, W, bias, out_m, N_m);
  k_text_wmma<<<dim3(gta), 128, 0, stream>>>(a_txt, W, bias, out_a, N_a);
}
